// CrossAttentionLayer_446676599069
// MI455X (gfx1250) — matrix operand order repaired and checked by reading, not compiled, not run on hardware
//
#include <hip/hip_runtime.h>
#include <hip/hip_bf16.h>

// ---------------------------------------------------------------------------
// MI455X (gfx1250) cross-attention: bf16 WMMA + flash attention.
// All operands kept bf16 row-major with contraction dim contiguous, so WMMA
// fragments are contiguous 16B/32B per-lane runs:
//   - A fragments load straight from global into VGPRs (global_load_b128 x2)
//   - B fragments stage via GLOBAL_LOAD_ASYNC_TO_LDS_B128 (ASYNCcnt),
//     double-buffered.
// Softmax row reductions use DPP quad_perm (xor1/2) + ds_swizzle (xor4/8).
// ---------------------------------------------------------------------------

typedef __attribute__((ext_vector_type(16))) __bf16 v16bf;
typedef __attribute__((ext_vector_type(8)))  __bf16 v8bf;
typedef __attribute__((ext_vector_type(4)))  __bf16 v4bf;
typedef __attribute__((ext_vector_type(8)))  float  v8f;

struct __align__(16) B16 { unsigned int x[4]; };
union Frag { v16bf v; unsigned int u[8]; };

#define D_MODEL 1024
#define HEADS   16
#define DEPTH   64

// --- CDNA5 async global->LDS copy (tracked by ASYNCcnt) --------------------
__device__ inline unsigned lds_off(const void* p) {
    // generic pointer to LDS: low 32 bits are the LDS byte offset
    return (unsigned)(unsigned long long)p;
}
__device__ inline void async_cp16(unsigned lds, const void* g) {
    asm volatile("global_load_async_to_lds_b128 %0, %1, off"
                 :: "v"(lds), "v"(g) : "memory");
}
__device__ inline void async_wait0() {
    asm volatile("s_wait_asynccnt 0" ::: "memory");
}

// --- cross-lane reductions within 16-lane rows (C-layout rows) -------------
template <int CTRL>
__device__ inline float dpp_xor(float x) {
    return __int_as_float(__builtin_amdgcn_update_dpp(
        0, __float_as_int(x), CTRL, 0xf, 0xf, true));
}
__device__ inline float swz_xor4(float x) {   // ds_swizzle xor-mask 4
    return __int_as_float(__builtin_amdgcn_ds_swizzle(__float_as_int(x), 0x101F));
}
__device__ inline float swz_xor8(float x) {   // ds_swizzle xor-mask 8
    return __int_as_float(__builtin_amdgcn_ds_swizzle(__float_as_int(x), 0x201F));
}
__device__ inline float rowmax16(float v) {
    v = fmaxf(v, dpp_xor<0xB1>(v));           // quad_perm(1,0,3,2) : xor 1
    v = fmaxf(v, dpp_xor<0x4E>(v));           // quad_perm(2,3,0,1) : xor 2
    v = fmaxf(v, swz_xor4(v));
    v = fmaxf(v, swz_xor8(v));
    return v;
}
__device__ inline float rowsum16(float v) {
    v += dpp_xor<0xB1>(v);
    v += dpp_xor<0x4E>(v);
    v += swz_xor4(v);
    v += swz_xor8(v);
    return v;
}

// ---------------------------------------------------------------------------
// fp32 -> bf16 elementwise convert (keeps layout)
// ---------------------------------------------------------------------------
__global__ __launch_bounds__(256) void conv_f32_bf16(
    const float* __restrict__ in, __bf16* __restrict__ out, int n4)
{
    int i = blockIdx.x * 256 + threadIdx.x;
    if (i >= n4) return;
    float4 f = ((const float4*)in)[i];
    v4bf p;
    p[0] = (__bf16)f.x; p[1] = (__bf16)f.y;
    p[2] = (__bf16)f.z; p[3] = (__bf16)f.w;
    ((v4bf*)out)[i] = p;
}

// ---------------------------------------------------------------------------
// GEMM:  Y[M,1024] = X[M,1024] @ W[1024,1024]^T + bias
// X bf16 row-major [M,K]; W bf16 row-major [N,K] (exactly B-frag friendly).
// 256 thr = 8 waves; block tile 128x64; A-frags direct global->VGPR;
// B tile async-staged to LDS, double-buffered.
// TRANS=true writes Y transposed as [N_total=1024, ldY] (for V^T).
// ---------------------------------------------------------------------------
template <typename TOut, bool TRANS>
__global__ __launch_bounds__(256) void gemm_wmma(
    const __bf16* __restrict__ X, const __bf16* __restrict__ W,
    const float* __restrict__ bias, TOut* __restrict__ Y, int ldY)
{
    const int K = D_MODEL;
    __shared__ v16bf Bs[2][4][32];     // [buf][ntile][lane]

    const int tid  = threadIdx.x;
    const int wave = tid >> 5;
    const int lane = tid & 31;
    const int m0 = blockIdx.y * 128;
    const int n0 = blockIdx.x * 64;
    const int lb = lane >> 4;

    // B staging: 4 frags x 32 lanes x 32B per k-step; 16B per thread.
    const int snt = (tid >> 6) & 3, slane = (tid >> 1) & 31, sh = tid & 1;
    const __bf16* bsrc = W + (size_t)(n0 + snt * 16 + (slane & 15)) * K
                           + (slane >> 4) * 8 + sh * 16;
    const unsigned bdst0     = lds_off((const char*)&Bs[0][snt][slane] + sh * 16);
    const unsigned bufstride = (unsigned)sizeof(Bs[0]);

    const __bf16* arow = X + (size_t)(m0 + wave * 16 + (lane & 15)) * K + lb * 8;

    v8f acc[4] = {};
    async_cp16(bdst0, bsrc);
    for (int k0 = 0, buf = 0; k0 < K; k0 += 32, buf ^= 1) {
        async_wait0();
        __syncthreads();
        if (k0 + 32 < K)
            async_cp16(bdst0 + (buf ^ 1) * bufstride, bsrc + k0 + 32);

        Frag a;
        *(B16*)&a.u[0] = *(const B16*)(arow + k0);        // k = k0 + 8*lb ..
        *(B16*)&a.u[4] = *(const B16*)(arow + k0 + 16);   // k = k0+16+8*lb ..
        #pragma unroll
        for (int nt = 0; nt < 4; ++nt)
            acc[nt] = __builtin_amdgcn_wmma_f32_16x16x32_bf16(
                false, a.v, false, Bs[buf][nt][lane], (short)0, acc[nt],
                false, false);
    }

    #pragma unroll
    for (int nt = 0; nt < 4; ++nt) {
        int col = n0 + nt * 16 + (lane & 15);
        float bv = bias[col];
        int rbase = m0 + wave * 16 + (lb << 3);
        if constexpr (TRANS) {
            v8bf pk;
            #pragma unroll
            for (int r = 0; r < 8; ++r) pk[r] = (__bf16)(acc[nt][r] + bv);
            *(v8bf*)((__bf16*)Y + (size_t)col * ldY + rbase) = pk;   // 16B store
        } else {
            #pragma unroll
            for (int r = 0; r < 8; ++r)
                Y[(size_t)(rbase + r) * ldY + col] = (TOut)(acc[nt][r] + bv);
        }
    }
}

// ---------------------------------------------------------------------------
// Flash cross-attention.  grid = (T/128, H, B), 256 thr = 8 waves.
// Wave owns 16 query rows (Q frags in registers).  32-key chunks:
// K [c,d] and Vt [d, b*C+c] tiles async-staged to LDS as B-frags
// (contiguous 32B per lane), double-buffered; frag f = wave id.
// ---------------------------------------------------------------------------
__global__ __launch_bounds__(256) void attn_flash_wmma(
    const __bf16* __restrict__ Qb, const __bf16* __restrict__ Kb,
    const __bf16* __restrict__ Vt, __bf16* __restrict__ Ob,
    int T, int C, int ldV)
{
    __shared__ v16bf KV[2][8][32];   // [buf][frag][lane]; 0-3: K^T, 4-7: V
    __shared__ v16bf Ps[8][32];      // per-wave P bounce (A-frag layout)

    const int tid  = threadIdx.x;
    const int wave = tid >> 5;
    const int lane = tid & 31;
    const int hh = blockIdx.y;
    const int b  = blockIdx.z;
    const int q0 = blockIdx.x * 128 + wave * 16;
    const int lb = lane >> 4;

    // Per-thread staging source: two 16B runs (k 8*lb.., 16+8*lb..) per chunk.
    const __bf16* src;
    size_t cstep;
    if (wave < 4) {                         // K frag: f = dstep*2 + ct
        int dstep = wave >> 1, ct = wave & 1;
        src = Kb + ((size_t)b * C + ct * 16 + (lane & 15)) * D_MODEL
                 + hh * DEPTH + dstep * 32 + lb * 8;
        cstep = (size_t)32 * D_MODEL;
    } else {                                // V frag: f = 4 + dtile
        int dt = wave - 4;
        src = Vt + (size_t)(hh * DEPTH + dt * 16 + (lane & 15)) * ldV
                 + (size_t)b * C + lb * 8;
        cstep = 32;
    }
    const unsigned dst0      = lds_off(&KV[0][wave][lane]);
    const unsigned bufstride = (unsigned)sizeof(KV[0]);

    // Q fragments: two contiguous 16B runs per lane per frag.
    const __bf16* qrow = Qb + ((size_t)b * T + q0 + (lane & 15)) * D_MODEL
                            + hh * DEPTH + lb * 8;
    Frag qa[2];
    #pragma unroll
    for (int kk = 0; kk < 2; ++kk) {
        *(B16*)&qa[kk].u[0] = *(const B16*)(qrow + kk * 32);
        *(B16*)&qa[kk].u[4] = *(const B16*)(qrow + kk * 32 + 16);
    }

    float mrow[8], lrow[8];
    v8f Oa[4] = {};
    #pragma unroll
    for (int r = 0; r < 8; ++r) { mrow[r] = -1e30f; lrow[r] = 0.0f; }
    const float scale = 0.125f;             // 1/sqrt(64)
    const int nchunk = C / 32;

    async_cp16(dst0, src);
    async_cp16(dst0 + 16, src + 16);
    for (int ci = 0, buf = 0; ci < nchunk; ++ci, buf ^= 1) {
        async_wait0();
        __syncthreads();
        if (ci + 1 < nchunk) {
            const __bf16* s2 = src + (size_t)(ci + 1) * cstep;
            async_cp16(dst0 + (buf ^ 1) * bufstride, s2);
            async_cp16(dst0 + (buf ^ 1) * bufstride + 16, s2 + 16);
        }

        // S = Q @ K^T   (2 n-tiles x 2 k-steps)
        v8f S[2] = {};
        #pragma unroll
        for (int nt = 0; nt < 2; ++nt) {
            S[nt] = __builtin_amdgcn_wmma_f32_16x16x32_bf16(
                false, qa[0].v, false, KV[buf][nt][lane],     (short)0, S[nt],
                false, false);
            S[nt] = __builtin_amdgcn_wmma_f32_16x16x32_bf16(
                false, qa[1].v, false, KV[buf][2 + nt][lane], (short)0, S[nt],
                false, false);
        }

        // Online softmax; DPP/ds_swizzle row reductions (no bpermute).
        #pragma unroll
        for (int r = 0; r < 8; ++r) {
            float s0 = S[0][r] * scale, s1 = S[1][r] * scale;
            float mx = rowmax16(fmaxf(s0, s1));
            float mnew  = fmaxf(mrow[r], mx);
            float alpha = __expf(mrow[r] - mnew);
            mrow[r] = mnew;
            float p0 = __expf(s0 - mnew);
            float p1 = __expf(s1 - mnew);
            float rs = rowsum16(p0 + p1);
            lrow[r] = lrow[r] * alpha + rs;
            #pragma unroll
            for (int t2 = 0; t2 < 4; ++t2) Oa[t2][r] *= alpha;

            // scatter P into per-wave A-frag layout
            int M = r + (lb << 3);
            int c = lane & 15;
            ((__bf16*)&Ps[wave][M + (((c >> 3) & 1) << 4)])
                [(c & 7) + ((c >> 4) << 3)] = (__bf16)p0;
            c = (lane & 15) + 16;
            ((__bf16*)&Ps[wave][M + (((c >> 3) & 1) << 4)])
                [(c & 7) + ((c >> 4) << 3)] = (__bf16)p1;
        }
        asm volatile("" ::: "memory");   // LDS is in-order per wave
        Frag pf; pf.v = Ps[wave][lane];

        // O += P @ V   (4 d-tiles, K-dim = 32 keys)
        #pragma unroll
        for (int t2 = 0; t2 < 4; ++t2)
            Oa[t2] = __builtin_amdgcn_wmma_f32_16x16x32_bf16(
                false, pf.v, false, KV[buf][4 + t2][lane], (short)0, Oa[t2],
                false, false);
    }

    // Normalize and write bf16 attention output [B,T,D] (head-interleaved).
    __bf16* obase = Ob + ((size_t)b * T + q0) * D_MODEL + hh * DEPTH;
    #pragma unroll
    for (int t2 = 0; t2 < 4; ++t2) {
        int col = t2 * 16 + (lane & 15);
        #pragma unroll
        for (int r = 0; r < 8; ++r) {
            int row = r + (lb << 3);
            obase[(size_t)row * D_MODEL + col] = (__bf16)(Oa[t2][r] / lrow[r]);
        }
    }
}

// ---------------------------------------------------------------------------
extern "C" void kernel_launch(void* const* d_in, const int* in_sizes, int n_in,
                              void* d_out, int out_size, void* d_ws, size_t ws_size,
                              hipStream_t stream)
{
    (void)in_sizes; (void)n_in; (void)out_size; (void)ws_size;
    const float* x   = (const float*)d_in[0];
    const float* ctx = (const float*)d_in[1];
    const float* Wq  = (const float*)d_in[2];
    const float* bq  = (const float*)d_in[3];
    const float* Wk  = (const float*)d_in[4];
    const float* bk  = (const float*)d_in[5];
    const float* Wv  = (const float*)d_in[6];
    const float* bv  = (const float*)d_in[7];
    const float* Wo  = (const float*)d_in[8];
    const float* bo  = (const float*)d_in[9];
    float* out = (float*)d_out;

    const int B = 2, T = 2048, C = 3072;
    const int MQ = B * T;      // 4096
    const int MC = B * C;      // 6144

    char* ws = (char*)d_ws;
    __bf16* xb  = (__bf16*)ws;  ws += (size_t)MQ * D_MODEL * 2;
    __bf16* cb  = (__bf16*)ws;  ws += (size_t)MC * D_MODEL * 2;
    __bf16* Wqb = (__bf16*)ws;  ws += (size_t)D_MODEL * D_MODEL * 2;
    __bf16* Wkb = (__bf16*)ws;  ws += (size_t)D_MODEL * D_MODEL * 2;
    __bf16* Wvb = (__bf16*)ws;  ws += (size_t)D_MODEL * D_MODEL * 2;
    __bf16* Wob = (__bf16*)ws;  ws += (size_t)D_MODEL * D_MODEL * 2;
    __bf16* Qb  = (__bf16*)ws;  ws += (size_t)MQ * D_MODEL * 2;
    __bf16* Kb  = (__bf16*)ws;  ws += (size_t)MC * D_MODEL * 2;
    __bf16* Vt  = (__bf16*)ws;  ws += (size_t)D_MODEL * MC * 2;   // [D, B*C]
    __bf16* Ab  = (__bf16*)ws;                                    // [B*T, D]

    // fp32 -> bf16 converts (layouts preserved; W stays [N,K] = B-frag order)
    conv_f32_bf16<<<(MQ * D_MODEL / 4 + 255) / 256, 256, 0, stream>>>(x,  xb,  MQ * D_MODEL / 4);
    conv_f32_bf16<<<(MC * D_MODEL / 4 + 255) / 256, 256, 0, stream>>>(ctx, cb, MC * D_MODEL / 4);
    conv_f32_bf16<<<(D_MODEL * D_MODEL / 4 + 255) / 256, 256, 0, stream>>>(Wq, Wqb, D_MODEL * D_MODEL / 4);
    conv_f32_bf16<<<(D_MODEL * D_MODEL / 4 + 255) / 256, 256, 0, stream>>>(Wk, Wkb, D_MODEL * D_MODEL / 4);
    conv_f32_bf16<<<(D_MODEL * D_MODEL / 4 + 255) / 256, 256, 0, stream>>>(Wv, Wvb, D_MODEL * D_MODEL / 4);
    conv_f32_bf16<<<(D_MODEL * D_MODEL / 4 + 255) / 256, 256, 0, stream>>>(Wo, Wob, D_MODEL * D_MODEL / 4);

    // Projections (Q, K row-major bf16; V written transposed [D, B*C]).
    gemm_wmma<__bf16, false><<<dim3(16, MQ / 128), 256, 0, stream>>>(xb, Wqb, bq, Qb, D_MODEL);
    gemm_wmma<__bf16, false><<<dim3(16, MC / 128), 256, 0, stream>>>(cb, Wkb, bk, Kb, D_MODEL);
    gemm_wmma<__bf16, true ><<<dim3(16, MC / 128), 256, 0, stream>>>(cb, Wvb, bv, Vt, MC);

    attn_flash_wmma<<<dim3(T / 128, HEADS, B), 256, 0, stream>>>(
        Qb, Kb, Vt, Ab, T, C, MC);

    // Output projection -> fp32
    gemm_wmma<float, false><<<dim3(16, MQ / 128), 256, 0, stream>>>(Ab, Wob, bo, out, D_MODEL);
}
